// SumCrossAttention_41240275976820
// MI455X (gfx1250) — matrix operand order repaired and checked by reading, not compiled, not run on hardware
//
#include <hip/hip_runtime.h>
#include <hip/hip_bf16.h>

// ---------------------------------------------------------------------------
// SumCrossAttention for MI455X (gfx1250, wave32, WMMA).
// All GEMMs (4 projections + QK^T + P*V) run on v_wmma_f32_16x16x32_bf16
// with f32 accumulation; softmax / mixing stay in f32 VALU.
// ---------------------------------------------------------------------------

typedef __attribute__((ext_vector_type(16))) __bf16 v16bf;
typedef __attribute__((ext_vector_type(8)))  float  v8f;

namespace {

constexpr int NB = 8;                 // batch
constexpr int NT = 243;               // t == t_sup
constexpr int NJ = 17;
constexpr int NH = 8;
constexpr int HD = 64;                // head dim
constexpr int DD = 512;               // model dim
constexpr int SP = 256;               // padded sequence (16 tiles of 16)
constexpr int MROWS = NB * NT * NJ;   // 33048 projection rows

union FragU { uint4 u[2]; v16bf v; };

__device__ __forceinline__ v8f wmma_bf16(v16bf a, v16bf b, v8f c) {
  // D = A(16x32) * B(32x16) + C, f32 accumulate
  return __builtin_amdgcn_wmma_f32_16x16x32_bf16(false, a, false, b,
                                                 (short)0, c, false, false);
}

// A fragment (16x32 bf16). rowp -> (row, k0) of a row-major MxK buffer.
// ISA 7.12.2: half 0 lanes hold K=half*8+{0..7} in v0..3, K=16+half*8+{0..7}
// in v4..7  => two b128 loads at byte offsets half*16 and 32+half*16.
__device__ __forceinline__ v16bf ldA_bf16(const __bf16* rowp, int hf) {
  const __bf16* p = rowp + hf * 8;
  FragU f;
  f.u[0] = *(const uint4*)(p);
  f.u[1] = *(const uint4*)(p + 16);
  return f.v;
}

__device__ __forceinline__ v16bf ldA_f32(const float* rowp, int hf) {
  const float* p = rowp + hf * 8;
  float4 x0 = *(const float4*)(p + 0);
  float4 x1 = *(const float4*)(p + 4);
  float4 x2 = *(const float4*)(p + 16);
  float4 x3 = *(const float4*)(p + 20);
  v16bf r;
  r[0]  = (__bf16)x0.x; r[1]  = (__bf16)x0.y; r[2]  = (__bf16)x0.z; r[3]  = (__bf16)x0.w;
  r[4]  = (__bf16)x1.x; r[5]  = (__bf16)x1.y; r[6]  = (__bf16)x1.z; r[7]  = (__bf16)x1.w;
  r[8]  = (__bf16)x2.x; r[9]  = (__bf16)x2.y; r[10] = (__bf16)x2.z; r[11] = (__bf16)x2.w;
  r[12] = (__bf16)x3.x; r[13] = (__bf16)x3.y; r[14] = (__bf16)x3.z; r[15] = (__bf16)x3.w;
  return r;
}

// B fragment (32x16) from an N x K row-major buffer (rowp -> (n, k0)).
// Lane half h holds K=h*8+{0..7} then K=16+h*8+{0..7} (same order as A).
__device__ __forceinline__ v16bf ldB_bf16(const __bf16* rowp, int hf) {
  const __bf16* p = rowp + hf * 8;
  FragU f;
  f.u[0] = *(const uint4*)(p);
  f.u[1] = *(const uint4*)(p + 16);
  return f.v;
}

__device__ __forceinline__ v16bf ldB_f32(const float* rowp, int hf) {
  const float* p = rowp + hf * 8;
  float4 x0 = *(const float4*)(p + 0);
  float4 x1 = *(const float4*)(p + 4);
  float4 x2 = *(const float4*)(p + 16);
  float4 x3 = *(const float4*)(p + 20);
  v16bf r;
  r[0]  = (__bf16)x0.x; r[1]  = (__bf16)x0.y; r[2]  = (__bf16)x0.z; r[3]  = (__bf16)x0.w;
  r[4]  = (__bf16)x1.x; r[5]  = (__bf16)x1.y; r[6]  = (__bf16)x1.z; r[7]  = (__bf16)x1.w;
  r[8]  = (__bf16)x2.x; r[9]  = (__bf16)x2.y; r[10] = (__bf16)x2.z; r[11] = (__bf16)x2.w;
  r[12] = (__bf16)x3.x; r[13] = (__bf16)x3.y; r[14] = (__bf16)x3.z; r[15] = (__bf16)x3.w;
  return r;
}

}  // namespace

// ---------------------------------------------------------------------------
// Generic projection GEMM: out[M,512] = A[M,512] @ W[512,512]^T (+ bias).
// Block = 128 threads (4 waves in 2x2), block tile 64x64, wave tile 32x32.
// ---------------------------------------------------------------------------
template <bool A_F32, bool FINAL>
__global__ __launch_bounds__(128) void proj_gemm_kernel(
    const void* __restrict__ Ap, const float* __restrict__ W,
    void* __restrict__ outp, const float* __restrict__ bias, int M) {
  const int lane = threadIdx.x & 31;
  const int wid  = threadIdx.x >> 5;
  const int hf   = lane >> 4;
  const int l16  = lane & 15;
  const int m0 = blockIdx.x * 64 + (wid & 1) * 32;
  const int n0 = blockIdx.y * 64 + (wid >> 1) * 32;

  int r0 = m0 + l16;      if (r0 > M - 1) r0 = M - 1;   // clamp padded rows
  int r1 = m0 + 16 + l16; if (r1 > M - 1) r1 = M - 1;
  const float*  Af = (const float*)Ap;
  const __bf16* Ab = (const __bf16*)Ap;
  const float* w0 = W + (size_t)(n0 + l16) * DD;
  const float* w1 = W + (size_t)(n0 + 16 + l16) * DD;

  v8f acc[2][2] = {};
  for (int ks = 0; ks < DD / 32; ++ks) {
    const int ko = ks * 32;
    v16bf a0, a1;
    if (A_F32) {
      a0 = ldA_f32(Af + (size_t)r0 * DD + ko, hf);
      a1 = ldA_f32(Af + (size_t)r1 * DD + ko, hf);
    } else {
      a0 = ldA_bf16(Ab + (size_t)r0 * DD + ko, hf);
      a1 = ldA_bf16(Ab + (size_t)r1 * DD + ko, hf);
    }
    v16bf b0 = ldB_f32(w0 + ko, hf);
    v16bf b1 = ldB_f32(w1 + ko, hf);
    acc[0][0] = wmma_bf16(a0, b0, acc[0][0]);
    acc[0][1] = wmma_bf16(a0, b1, acc[0][1]);
    acc[1][0] = wmma_bf16(a1, b0, acc[1][0]);
    acc[1][1] = wmma_bf16(a1, b1, acc[1][1]);
  }

  // C layout: M = v + 8*half, N = lane%16
#pragma unroll
  for (int mi = 0; mi < 2; ++mi)
#pragma unroll
    for (int ni = 0; ni < 2; ++ni)
#pragma unroll
      for (int v = 0; v < 8; ++v) {
        const int r = m0 + mi * 16 + v + hf * 8;
        const int c = n0 + ni * 16 + l16;
        if (r < M) {
          if (FINAL)
            ((float*)outp)[(size_t)r * DD + c] = acc[mi][ni][v] + bias[c];
          else
            ((__bf16*)outp)[(size_t)r * DD + c] = (__bf16)acc[mi][ni][v];
        }
      }
}

// ---------------------------------------------------------------------------
// Build Vt[b][h][j][hd=64][s=256] (bf16, s padded with zeros) so the P*V
// B-fragments become two contiguous 16B reads per lane.
// ---------------------------------------------------------------------------
__global__ __launch_bounds__(256) void vtrans_kernel(
    const __bf16* __restrict__ Vp, __bf16* __restrict__ Vt) {
  const int bhj = blockIdx.x;
  const int jj = bhj % NJ;
  const int hh = (bhj / NJ) % NH;
  const int bb = bhj / (NJ * NH);
  __bf16* out = Vt + ((size_t)(bb * NH + hh) * NJ + jj) * HD * SP;
  const __bf16* src =
      Vp + (size_t)bb * NT * NJ * DD + (size_t)jj * DD + hh * HD;
  for (int idx = threadIdx.x; idx < HD * SP; idx += 256) {
    const int d = idx >> 8;     // head-dim row
    const int s = idx & 255;    // sequence column
    __bf16 v = (__bf16)0.0f;
    if (s < NT) v = src[(size_t)s * NJ * DD + d];
    out[idx] = v;
  }
}

// ---------------------------------------------------------------------------
// Attention: one block per (b,h,j); 4 waves, each wave owns 16-row t-tiles.
// scores(16x256) in accumulators -> softmax -> mix attmap -> softmax ->
// bf16 probs via LDS strip -> P*V (16x64) -> bf16 O.
// ---------------------------------------------------------------------------
__global__ __launch_bounds__(128) void attn_kernel(
    const __bf16* __restrict__ Qp, const __bf16* __restrict__ Kp,
    const __bf16* __restrict__ Vt, const float* __restrict__ attmap,
    const float* __restrict__ weightp, __bf16* __restrict__ O) {
  __shared__ alignas(16) __bf16 sA[4][16][264];  // per-wave prob strip

  const int lane = threadIdx.x & 31;
  const int wid  = threadIdx.x >> 5;
  const int hf   = lane >> 4;
  const int l16  = lane & 15;

  const int bhj = blockIdx.x;
  const int jj = bhj % NJ;
  const int hh = (bhj / NJ) % NH;
  const int bb = bhj / (NJ * NH);

  const float w  = weightp[0];
  const float wc = 1.0f - w;
  const float scl = 0.125f;  // hd^-0.5 = 1/8

  const size_t rstride = (size_t)NJ * DD;  // stride between t rows
  const __bf16* Qb =
      Qp + (size_t)bb * NT * NJ * DD + (size_t)jj * DD + hh * HD;
  const __bf16* Kb =
      Kp + (size_t)bb * NT * NJ * DD + (size_t)jj * DD + hh * HD;
  const __bf16* Vb = Vt + ((size_t)(bb * NH + hh) * NJ + jj) * HD * SP;
  const float* Amap = attmap + ((size_t)(bb * NH + hh) * NJ + jj) * NT * NT;

  for (int tt = wid; tt < 16; tt += 4) {
    const int t0 = tt * 16;
    int qrow = t0 + l16;
    if (qrow > NT - 1) qrow = NT - 1;
    const v16bf aq0 = ldA_bf16(Qb + (size_t)qrow * rstride, hf);       // k 0..31
    const v16bf aq1 = ldA_bf16(Qb + (size_t)qrow * rstride + 32, hf);  // k 32..63

    // ---- phase 1: scores = Q K^T, 16 s-tiles held in accumulators -------
    v8f sc[16];
#pragma unroll
    for (int st = 0; st < 16; ++st) {
      int srow = st * 16 + l16;
      if (srow > NT - 1) srow = NT - 1;
      const v16bf b0 = ldB_bf16(Kb + (size_t)srow * rstride, hf);
      const v16bf b1 = ldB_bf16(Kb + (size_t)srow * rstride + 32, hf);
      v8f c = {0.f, 0.f, 0.f, 0.f, 0.f, 0.f, 0.f, 0.f};
      c = wmma_bf16(aq0, b0, c);
      c = wmma_bf16(aq1, b1, c);
      sc[st] = c;
    }

    // ---- softmax #1 (row-wise over s; rows live in half-wave lanes) -----
    float rmax[8], rsum[8];
#pragma unroll
    for (int v = 0; v < 8; ++v) rmax[v] = -3.0e38f;
#pragma unroll
    for (int st = 0; st < 16; ++st) {
      const bool valid = (st * 16 + l16) < NT;
#pragma unroll
      for (int v = 0; v < 8; ++v) {
        const float x = valid ? sc[st][v] * scl : -3.0e38f;
        sc[st][v] = x;
        rmax[v] = fmaxf(rmax[v], x);
      }
    }
#pragma unroll
    for (int v = 0; v < 8; ++v) {
#pragma unroll
      for (int mm = 1; mm < 16; mm <<= 1)
        rmax[v] = fmaxf(rmax[v], __shfl_xor(rmax[v], mm, 32));
      rsum[v] = 0.0f;
    }
#pragma unroll
    for (int st = 0; st < 16; ++st)
#pragma unroll
      for (int v = 0; v < 8; ++v) {
        const float e = __expf(sc[st][v] - rmax[v]);
        sc[st][v] = e;
        rsum[v] += e;
      }
#pragma unroll
    for (int v = 0; v < 8; ++v) {
#pragma unroll
      for (int mm = 1; mm < 16; mm <<= 1)
        rsum[v] += __shfl_xor(rsum[v], mm, 32);
      rsum[v] = 1.0f / rsum[v];
    }

    // ---- mix with attmap, re-mask ---------------------------------------
#pragma unroll
    for (int st = 0; st < 16; ++st) {
      const int scol = st * 16 + l16;
      const bool valid = scol < NT;
#pragma unroll
      for (int v = 0; v < 8; ++v) {
        int trow = t0 + v + hf * 8;
        if (trow > NT - 1) trow = NT - 1;
        const float am = valid ? Amap[(size_t)trow * NT + scol] : 0.0f;
        const float x = w * (sc[st][v] * rsum[v]) + wc * am;
        sc[st][v] = valid ? x : -3.0e38f;
      }
    }

    // ---- softmax #2 ------------------------------------------------------
#pragma unroll
    for (int v = 0; v < 8; ++v) rmax[v] = -3.0e38f;
#pragma unroll
    for (int st = 0; st < 16; ++st)
#pragma unroll
      for (int v = 0; v < 8; ++v) rmax[v] = fmaxf(rmax[v], sc[st][v]);
#pragma unroll
    for (int v = 0; v < 8; ++v) {
#pragma unroll
      for (int mm = 1; mm < 16; mm <<= 1)
        rmax[v] = fmaxf(rmax[v], __shfl_xor(rmax[v], mm, 32));
      rsum[v] = 0.0f;
    }
#pragma unroll
    for (int st = 0; st < 16; ++st)
#pragma unroll
      for (int v = 0; v < 8; ++v) {
        const float e = __expf(sc[st][v] - rmax[v]);
        sc[st][v] = e;
        rsum[v] += e;
      }
#pragma unroll
    for (int v = 0; v < 8; ++v) {
#pragma unroll
      for (int mm = 1; mm < 16; mm <<= 1)
        rsum[v] += __shfl_xor(rsum[v], mm, 32);
      rsum[v] = 1.0f / rsum[v];
    }

    // ---- probs (bf16) to this wave's LDS strip --------------------------
#pragma unroll
    for (int st = 0; st < 16; ++st)
#pragma unroll
      for (int v = 0; v < 8; ++v)
        sA[wid][v + hf * 8][st * 16 + l16] = (__bf16)(sc[st][v] * rsum[v]);

    // ---- phase 2: O = P V  (16x64 output, K = 256) ----------------------
    v8f oacc[4] = {};
#pragma unroll
    for (int ks = 0; ks < 8; ++ks) {
      const v16bf a = ldA_bf16(&sA[wid][l16][ks * 32], hf);  // ds_load_b128
#pragma unroll
      for (int nt = 0; nt < 4; ++nt) {
        const v16bf b =
            ldB_bf16(Vb + (size_t)(nt * 16 + l16) * SP + ks * 32, hf);
        oacc[nt] = wmma_bf16(a, b, oacc[nt]);
      }
    }

#pragma unroll
    for (int nt = 0; nt < 4; ++nt)
#pragma unroll
      for (int v = 0; v < 8; ++v) {
        const int trow = t0 + v + hf * 8;
        if (trow < NT)
          O[(((size_t)bb * NT + trow) * NJ + jj) * DD + hh * HD + nt * 16 +
            l16] = (__bf16)oacc[nt][v];
      }
  }
}

// ---------------------------------------------------------------------------
extern "C" void kernel_launch(void* const* d_in, const int* in_sizes, int n_in,
                              void* d_out, int out_size, void* d_ws,
                              size_t ws_size, hipStream_t stream) {
  (void)in_sizes; (void)n_in; (void)out_size; (void)ws_size;
  const float* q      = (const float*)d_in[0];
  const float* kv     = (const float*)d_in[1];
  const float* attmap = (const float*)d_in[2];
  const float* weight = (const float*)d_in[3];
  const float* Wq     = (const float*)d_in[4];
  const float* Wk     = (const float*)d_in[5];
  const float* Wv     = (const float*)d_in[6];
  const float* Wp     = (const float*)d_in[7];
  const float* bp     = (const float*)d_in[8];

  // Workspace layout (bf16): Qp | Kp | Vp | O | Vt   (~171 MB total)
  char* ws = (char*)d_ws;
  const size_t SZP = (size_t)MROWS * DD * sizeof(__bf16);  // 33.8 MB each
  __bf16* Qp = (__bf16*)(ws + 0 * SZP);
  __bf16* Kp = (__bf16*)(ws + 1 * SZP);
  __bf16* Vp = (__bf16*)(ws + 2 * SZP);
  __bf16* Ob = (__bf16*)(ws + 3 * SZP);
  __bf16* Vt = (__bf16*)(ws + 4 * SZP);

  const dim3 pgrid((MROWS + 63) / 64, DD / 64);
  proj_gemm_kernel<true, false><<<pgrid, 128, 0, stream>>>(q,  Wq, Qp, nullptr, MROWS);
  proj_gemm_kernel<true, false><<<pgrid, 128, 0, stream>>>(kv, Wk, Kp, nullptr, MROWS);
  proj_gemm_kernel<true, false><<<pgrid, 128, 0, stream>>>(kv, Wv, Vp, nullptr, MROWS);

  vtrans_kernel<<<NB * NH * NJ, 256, 0, stream>>>(Vp, Vt);

  attn_kernel<<<NB * NH * NJ, 128, 0, stream>>>(Qp, Kp, Vt, attmap, weight, Ob);

  proj_gemm_kernel<false, true><<<pgrid, 128, 0, stream>>>(Ob, Wp, (float*)d_out, bp, MROWS);
}
